// Net_64415919506077
// MI455X (gfx1250) — matrix operand order repaired and checked by reading, not compiled, not run on hardware
//
#include <hip/hip_runtime.h>
#include <hip/hip_bf16.h>
#include <math.h>

// ---------------------------------------------------------------------------
// Problem constants (match reference)
// ---------------------------------------------------------------------------
#define NB    256        // graphs
#define NPG   512        // nodes per graph (all graphs full -> mask all true)
#define NTOT  (NB*NPG)   // 131072 nodes
#define NE    2097152    // edges
#define CIN   128
#define HDIM  64
#define K1    256
#define K2    128
#define NOUT  10

typedef __attribute__((ext_vector_type(16))) _Float16 v16h;
typedef __attribute__((ext_vector_type(8)))  float    v8f;

// ---------------------------------------------------------------------------
// Batched WMMA GEMM with LDS staging.
//   C[z] = act( accum*C[z] + op(A[z]) @ B[z] + bias )
// Block = 128 threads (4 waves) computing a 64x64 tile of C.
// Wave w owns rows [64*by + 16w, +16) and all four 16x16 column tiles.
// K stepped by 32. Requirements (all satisfied by call sites):
//   M % 64 == 0, N % 64 == 0, K % 32 == 0, pointers 16B aligned.
// LDS layout:
//   Alds: fragment-linear, addr = (rowblk*32 + lane)*16 + slot, where per the
//         CDNA5 16-bit A layout lane = (m&15) + 16*k[3],
//         slot = k[0] + 2*k[2:1] + 8*k[4]  -> each lane's 16 slots contiguous.
//   Blds: [n][k] (n-major), so lane reads k-contiguous 16-half chunks,
//         per the CDNA5 16-bit B layout (lane = n&15 (+16 for k[3]=1)).
// Fragments then load as single 32-byte LDS reads.
// ---------------------------------------------------------------------------
template <int TA>
__global__ void __launch_bounds__(128)
k_wmma_gemm(const float* __restrict__ A, const float* __restrict__ Bm,
            float* __restrict__ C, const float* __restrict__ bias,
            int M, int N, int K,
            long long sA, long long sB, long long sC,
            int act, int accum) {
  __shared__ _Float16 Alds[64 * 32];   // 4 KB
  __shared__ _Float16 Blds[64 * 32];   // 4 KB
  const float* Ab = A  + (long long)blockIdx.z * sA;
  const float* Bb = Bm + (long long)blockIdx.z * sB;
  float*       Cb = C  + (long long)blockIdx.z * sC;
  const int t    = threadIdx.x;
  const int lane = t & 31;
  const int wave = t >> 5;
  const int row0 = blockIdx.y * 64;
  const int col0 = blockIdx.x * 64;
  const long long lda = TA ? (long long)M : (long long)K;

  v8f acc[4] = {};
  for (int k0 = 0; k0 < K; k0 += 32) {
    __syncthreads();                       // previous-iteration reads done
    if (TA) {
      // stored A is K x M; tile rows k (32) x cols m (64), coalesced over m
      int kk = t >> 4, mq = (t & 15) << 2;
      for (int k = kk; k < 32; k += 8) {
        float4 v = *(const float4*)(Ab + (long long)(k0 + k) * lda + row0 + mq);
        int j  = (k & 1) + ((k >> 1) & 3) * 2 + (k >> 4) * 8;
        int bs = ((mq >> 4) * 32 + ((k >> 3) & 1) * 16 + (mq & 15)) * 16 + j;
        Alds[bs]      = (_Float16)v.x;     // consecutive m -> +16 halves
        Alds[bs + 16] = (_Float16)v.y;
        Alds[bs + 32] = (_Float16)v.z;
        Alds[bs + 48] = (_Float16)v.w;
      }
    } else {
      // stored A is M x K; tile rows m (64) x cols k (32), coalesced over k
      int mm0 = t >> 3, kq = (t & 7) << 2;
      int j  = (kq & 1) + ((kq >> 1) & 3) * 2 + (kq >> 4) * 8;  // 4 contiguous slots
      int lh = ((kq >> 3) & 1) * 16;
      for (int mm = mm0; mm < 64; mm += 16) {
        float4 v = *(const float4*)(Ab + (long long)(row0 + mm) * lda + k0 + kq);
        _Float16* dst = &Alds[((mm >> 4) * 32 + lh + (mm & 15)) * 16 + j];
        dst[0] = (_Float16)v.x; dst[1] = (_Float16)v.y;
        dst[2] = (_Float16)v.z; dst[3] = (_Float16)v.w;
      }
    }
    {
      // B is K x N; tile rows k (32) x cols n (64), coalesced over n
      int kk = t >> 4, nq = (t & 15) << 2;
      if (k0 + 32 < K)                      // prefetch next K-strip of B
        __builtin_prefetch(Bb + (long long)(k0 + 32 + kk) * N + col0 + nq, 0, 1);
      for (int k = kk; k < 32; k += 8) {
        float4 v = *(const float4*)(Bb + (long long)(k0 + k) * N + col0 + nq);
        int bs = nq * 32 + ((k >> 3) & 1) * 16 + (k & 7) + (k >> 4) * 8;  // [n][k in fragment (table) order], consecutive n -> +32 halves
        Blds[bs]      = (_Float16)v.x;
        Blds[bs + 32] = (_Float16)v.y;
        Blds[bs + 64] = (_Float16)v.z;
        Blds[bs + 96] = (_Float16)v.w;
      }
    }
    __syncthreads();
    v16h a = *(const v16h*)&Alds[(wave * 32 + lane) * 16];
#pragma unroll
    for (int c = 0; c < 4; ++c) {
      v16h b = *(const v16h*)&Blds[(c * 16 + (lane & 15)) * 32 + (lane >> 4) * 16];
      acc[c] = __builtin_amdgcn_wmma_f32_16x16x32_f16(false, a, false, b, (short)0,
                                                      acc[c], false, false);
    }
  }

  // epilogue
  int nn = lane & 15;
  int mb = row0 + wave * 16 + ((lane >> 4) << 3);
#pragma unroll
  for (int c = 0; c < 4; ++c) {
    int ncol = col0 + c * 16 + nn;
#pragma unroll
    for (int v = 0; v < 8; ++v) {
      long long ci = (long long)(mb + v) * N + ncol;
      float r = acc[c][v];
      if (accum) r += Cb[ci];
      if (bias)  r += bias[ncol];
      if (act == 1) r = fmaxf(r, 0.f);
      else if (act == 2) {                  // selu
        const float sc = 1.0507009873554805f, al = 1.6732632423543772f;
        r = (r > 0.f) ? sc * r : sc * al * (__expf(r) - 1.f);
      }
      Cb[ci] = r;
    }
  }
}

// ---------------------------------------------------------------------------
// Elementwise / irregular helpers
// ---------------------------------------------------------------------------
__global__ void k_fill(float* p, float v, long long n) {
  long long i = (long long)blockIdx.x * blockDim.x + threadIdx.x;
  long long st = (long long)gridDim.x * blockDim.x;
  for (; i < n; i += st) p[i] = v;
}

__global__ void k_deg(const int* __restrict__ ei, float* __restrict__ deg) {
  long long e = (long long)blockIdx.x * blockDim.x + threadIdx.x;
  if (e >= NE) return;
  atomicAdd(deg + ei[NE + e], 1.0f);             // dst in-degree (+1 self loop preset)
}

__global__ void k_dinv(const float* __restrict__ deg, float* __restrict__ dinv) {
  long long i = (long long)blockIdx.x * blockDim.x + threadIdx.x;
  if (i < NTOT) dinv[i] = rsqrtf(deg[i]);
}

// out[dst] += h0[src] * dinv[src]*dinv[dst]; 16 threads/edge, 4 channels each
__global__ void k_scatter(const int* __restrict__ ei, const float* __restrict__ h0,
                          const float* __restrict__ dinv, float* __restrict__ out) {
  long long idx = (long long)blockIdx.x * blockDim.x + threadIdx.x;
  if (idx >= (long long)NE * 16) return;
  long long e = idx >> 4;
  int c4 = (int)(idx & 15) * 4;
  int s = ei[e], d = ei[NE + e];
  float cf = dinv[s] * dinv[d];
  const float4 hv = *reinterpret_cast<const float4*>(h0 + (long long)s * HDIM + c4);
  float* o = out + (long long)d * HDIM + c4;
  atomicAdd(o + 0, hv.x * cf);
  atomicAdd(o + 1, hv.y * cf);
  atomicAdd(o + 2, hv.z * cf);
  atomicAdd(o + 3, hv.w * cf);
}

// h = relu(agg + h0*dinv^2 + bias)   (writes in place over agg)
__global__ void k_gcn_final(float* __restrict__ agg, const float* __restrict__ h0,
                            const float* __restrict__ dinv, const float* __restrict__ b) {
  long long idx = (long long)blockIdx.x * blockDim.x + threadIdx.x;
  if (idx >= (long long)NTOT * HDIM) return;
  long long i = idx >> 6;  int c = (int)(idx & 63);
  float di = dinv[i];
  agg[idx] = fmaxf(agg[idx] + h0[idx] * di * di + b[c], 0.f);
}

__global__ void k_adj_build(const int* __restrict__ ei, float* __restrict__ adj) {
  long long e = (long long)blockIdx.x * blockDim.x + threadIdx.x;
  if (e >= NE) return;
  int s = ei[e], d = ei[NE + e];
  long long g = s >> 9;
  atomicAdd(adj + (g << 18) + (long long)(s & 511) * NPG + (d & 511), 1.0f);
}

// row-wise softmax in place (cols <= 256)
__global__ void k_softmax(float* __restrict__ x, int cols) {
  long long row = blockIdx.x;
  float* p = x + row * cols;
  int t = threadIdx.x;
  __shared__ float sm[256];
  float v = (t < cols) ? p[t] : -1e30f;
  sm[t] = v; __syncthreads();
  for (int s = 128; s > 0; s >>= 1) { if (t < s) sm[t] = fmaxf(sm[t], sm[t+s]); __syncthreads(); }
  float mx = sm[0]; __syncthreads();
  float e = (t < cols) ? __expf(v - mx) : 0.f;
  sm[t] = e; __syncthreads();
  for (int s = 128; s > 0; s >>= 1) { if (t < s) sm[t] += sm[t+s]; __syncthreads(); }
  if (t < cols) p[t] = e / sm[0];
}

// out[row] = sum(in[row, :cols]);  one wave per row
__global__ void k_rowsum(const float* __restrict__ in, float* __restrict__ out,
                         long long rows, int cols) {
  long long row = (long long)blockIdx.x * 4 + (threadIdx.x >> 5);
  int lane = threadIdx.x & 31;
  if (row >= rows) return;
  const float* p = in + row * cols;
  float s = 0.f;
  for (int j = lane; j < cols; j += 32) s += p[j];
  for (int m = 16; m; m >>= 1) s += __shfl_xor(s, m, 32);
  if (!lane) out[row] = s;
}

// ca[b,k] = sum_n s[b,n,k]*degd[b,n];  cs[b,k] = sum_n s[b,n,k]
__global__ void k_ca_colsum(const float* __restrict__ s, const float* __restrict__ degd,
                            float* __restrict__ ca, float* __restrict__ cs,
                            int Npg, int K) {
  int idx = blockIdx.x * blockDim.x + threadIdx.x;
  if (idx >= NB * K) return;
  int b = idx / K, k = idx - b * K;
  const float* sp = s + (long long)b * Npg * K + k;
  const float* dp = degd + (long long)b * Npg;
  float a = 0.f, c = 0.f;
  for (int n = 0; n < Npg; ++n) { float sv = sp[(long long)n * K]; a += sv * dp[n]; c += sv; }
  ca[idx] = a; cs[idx] = c;
}

// per-batch trace and sum-of-squares of KxK matrix
__global__ void k_mat_stats(const float* __restrict__ m, int K,
                            float* __restrict__ tr, float* __restrict__ ssq) {
  int b = blockIdx.x, t = threadIdx.x;
  const float* p = m + (long long)b * K * K;
  float sq = 0.f, dg = 0.f;
  for (int i = t; i < K * K; i += 256) {
    float v = p[i];
    sq += v * v;
    if (i / K == i % K) dg += v;
  }
  __shared__ float a0[256], a1[256];
  a0[t] = sq; a1[t] = dg; __syncthreads();
  for (int s = 128; s; s >>= 1) { if (t < s) { a0[t] += a0[t+s]; a1[t] += a1[t+s]; } __syncthreads(); }
  if (!t) { ssq[b] = a0[0]; tr[b] = a1[0]; }
}

// spectral + ortho + cluster losses, accumulated into *loss (mean over NB)
__global__ void k_losses(const float* __restrict__ trA, const float* __restrict__ ssq,
                         const float* __restrict__ sstr, const float* __restrict__ cs,
                         const float* __restrict__ ca, const float* __restrict__ twom,
                         int K, float msum, float* __restrict__ loss) {
  int b = blockIdx.x * blockDim.x + threadIdx.x;
  if (b >= NB) return;
  float caq = 0.f, csq = 0.f;
  for (int k = 0; k < K; ++k) {
    float c = ca[b * K + k]; caq += c * c;
    float x = cs[b * K + k]; csq += x * x;
  }
  float tm = twom[b];
  float sk = sqrtf((float)K);
  float sp = -(trA[b] - caq / tm) / tm;
  float fro = sqrtf(ssq[b]);
  // ||ss/fro - I/sqrt(K)||_F = sqrt(2 - 2*tr(ss)/(fro*sqrt(K)))
  float ortho = sqrtf(fmaxf(0.f, 2.f - 2.f * sstr[b] / (fro * sk)));
  float clus = sqrtf(csq) / msum * sk - 1.f;
  atomicAdd(loss, (sp + ortho + clus) * (1.0f / NB));
}

// a = normalize(zero_diag(oa)):  a[i,j] = oa[i,j]/(d[i]*d[j]), d=sqrt(rowsum)+eps
__global__ void k_finalize_a(const float* __restrict__ oa, float* __restrict__ a, int K) {
  int b = blockIdx.x, t = threadIdx.x;
  const float* p = oa + (long long)b * K * K;
  float*       q = a  + (long long)b * K * K;
  __shared__ float d[256];
  if (t < K) {
    float s = 0.f;
    for (int j = 0; j < K; ++j) if (j != t) s += p[t * K + j];
    d[t] = sqrtf(s) + 1e-15f;
  }
  __syncthreads();
  for (int i = t; i < K * K; i += 256) {
    int r = i / K, c = i - r * K;
    q[i] = (r == c) ? 0.f : p[i] / (d[r] * d[c]);
  }
}

// g[b,c] = mean_k x4[b,k,c]
__global__ void k_mean(const float* __restrict__ x4, float* __restrict__ g, int Kc) {
  int idx = blockIdx.x * blockDim.x + threadIdx.x;
  if (idx >= NB * HDIM) return;
  int b = idx >> 6, c = idx & 63;
  const float* p = x4 + (long long)b * Kc * HDIM + c;
  float s = 0.f;
  for (int k = 0; k < Kc; ++k) s += p[(long long)k * HDIM];
  g[idx] = s / (float)Kc;
}

// final linear (64->10) + log_softmax, one thread per graph
__global__ void k_head(const float* __restrict__ g1, const float* __restrict__ w,
                       const float* __restrict__ bias, float* __restrict__ out) {
  int b = blockIdx.x * blockDim.x + threadIdx.x;
  if (b >= NB) return;
  float lg[NOUT];
  float mx = -1e30f;
  for (int o = 0; o < NOUT; ++o) {
    float s = bias[o];
    for (int c = 0; c < HDIM; ++c) s += g1[b * HDIM + c] * w[c * NOUT + o];
    lg[o] = s; mx = fmaxf(mx, s);
  }
  float se = 0.f;
  for (int o = 0; o < NOUT; ++o) se += __expf(lg[o] - mx);
  float lse = mx + __logf(se);
  for (int o = 0; o < NOUT; ++o) out[b * NOUT + o] = lg[o] - lse;
}

// ---------------------------------------------------------------------------
// Host launcher
// ---------------------------------------------------------------------------
extern "C" void kernel_launch(void* const* d_in, const int* in_sizes, int n_in,
                              void* d_out, int out_size, void* d_ws, size_t ws_size,
                              hipStream_t stream) {
  const float* x       = (const float*)d_in[0];
  const int*   ei      = (const int*)  d_in[1];
  const float* gcn_w   = (const float*)d_in[3];
  const float* gcn_b   = (const float*)d_in[4];
  const float* p1_w1   = (const float*)d_in[5];
  const float* p1_b1   = (const float*)d_in[6];
  const float* p1_w2   = (const float*)d_in[7];
  const float* p1_b2   = (const float*)d_in[8];
  const float* c2_wrel = (const float*)d_in[9];
  const float* c2_brel = (const float*)d_in[10];
  const float* c2_wroot= (const float*)d_in[11];
  const float* p2_w1   = (const float*)d_in[12];
  const float* p2_b1   = (const float*)d_in[13];
  const float* p2_w2   = (const float*)d_in[14];
  const float* p2_b2   = (const float*)d_in[15];
  const float* c3_wrel = (const float*)d_in[16];
  const float* c3_brel = (const float*)d_in[17];
  const float* c3_wroot= (const float*)d_in[18];
  const float* lin1_w  = (const float*)d_in[19];
  const float* lin1_b  = (const float*)d_in[20];
  const float* lin2_w  = (const float*)d_in[21];
  const float* lin2_b  = (const float*)d_in[22];
  float* out = (float*)d_out;

  // workspace arena (floats); total ~ 263M floats (~1.05 GB)
  float* W = (float*)d_ws;
  size_t off = 0;
  auto AL = [&](size_t n) { size_t r = off; off += (n + 63) & ~size_t(63); return W + r; };
  float* deg   = AL(NTOT);
  float* dinv  = AL(NTOT);
  float* h0    = AL((size_t)NTOT * HDIM);
  float* xd    = AL((size_t)NTOT * HDIM);           // agg -> final node features
  float* adj   = AL((size_t)NB * NPG * NPG);        // 67.1M
  float* s1    = AL((size_t)NTOT * K1);             // logits then softmax (in place)
  float* t1    = AL((size_t)NTOT * HDIM);
  float* T1    = AL((size_t)NB * NPG * K1);         // adj @ s1
  float* oa1   = AL((size_t)NB * K1 * K1);
  float* a1    = AL((size_t)NB * K1 * K1);
  float* ss1   = AL((size_t)NB * K1 * K1);
  float* x1    = AL((size_t)NB * K1 * HDIM);
  float* degd1 = AL((size_t)NB * NPG);
  float* twom1 = AL(NB);
  float* ca1   = AL((size_t)NB * K1);
  float* cs1   = AL((size_t)NB * K1);
  float* trA1  = AL(NB); float* trash1 = AL(NB);
  float* ssq1  = AL(NB); float* sstr1  = AL(NB);
  float* u1    = AL((size_t)NB * K1 * HDIM);
  float* x2    = AL((size_t)NB * K1 * HDIM);
  float* t2    = AL((size_t)NB * K1 * HDIM);
  float* s2    = AL((size_t)NB * K1 * K2);
  float* T2    = AL((size_t)NB * K1 * K2);
  float* oa2   = AL((size_t)NB * K2 * K2);
  float* a2    = AL((size_t)NB * K2 * K2);
  float* ss2   = AL((size_t)NB * K2 * K2);
  float* x3    = AL((size_t)NB * K2 * HDIM);
  float* degd2 = AL((size_t)NB * K1);
  float* twom2 = AL(NB);
  float* ca2   = AL((size_t)NB * K2);
  float* cs2   = AL((size_t)NB * K2);
  float* trA2  = AL(NB); float* trash2 = AL(NB);
  float* ssq2  = AL(NB); float* sstr2  = AL(NB);
  float* u2    = AL((size_t)NB * K2 * HDIM);
  float* x4    = AL((size_t)NB * K2 * HDIM);
  float* g     = AL((size_t)NB * HDIM);
  float* g1    = AL((size_t)NB * HDIM);
  (void)ws_size; (void)n_in; (void)in_sizes; (void)out_size;

  auto fill = [&](float* p, float v, long long n) {
    int blocks = (int)((n + 255) / 256); if (blocks > 262144) blocks = 262144;
    k_fill<<<blocks, 256, 0, stream>>>(p, v, n);
  };
  auto gemm = [&](int TA, const float* A, const float* Bm, float* C, const float* bias,
                  int M, int N, int K, int batch,
                  long long sA, long long sB, long long sC, int act, int accum) {
    dim3 grd(N / 64, M / 64, batch);
    if (TA) k_wmma_gemm<1><<<grd, 128, 0, stream>>>(A, Bm, C, bias, M, N, K, sA, sB, sC, act, accum);
    else    k_wmma_gemm<0><<<grd, 128, 0, stream>>>(A, Bm, C, bias, M, N, K, sA, sB, sC, act, accum);
  };

  // ---- loss accumulator ----
  fill(out + NB * NOUT, 0.f, 1);

  // ---- GCN conv: h = relu(D^-1/2 (A+I) D^-1/2 (x @ w) + b) ----
  fill(deg, 1.0f, NTOT);                                        // self loops
  k_deg<<<(NE + 255) / 256, 256, 0, stream>>>(ei, deg);
  k_dinv<<<(NTOT + 255) / 256, 256, 0, stream>>>(deg, dinv);
  gemm(0, x, gcn_w, h0, nullptr, NTOT, HDIM, CIN, 1, 0, 0, 0, 0, 0);
  fill(xd, 0.f, (long long)NTOT * HDIM);
  k_scatter<<<(int)(((long long)NE * 16 + 255) / 256), 256, 0, stream>>>(ei, h0, dinv, xd);
  k_gcn_final<<<(int)(((long long)NTOT * HDIM + 255) / 256), 256, 0, stream>>>(xd, h0, dinv, gcn_b);

  // ---- dense adjacency (graphs are full & contiguous: pos = node & 511) ----
  fill(adj, 0.f, (long long)NB * NPG * NPG);
  k_adj_build<<<(NE + 255) / 256, 256, 0, stream>>>(ei, adj);

  // ---- DMoN pool 1 ----
  gemm(0, xd, p1_w1, t1, p1_b1, NTOT, HDIM, HDIM, 1, 0, 0, 0, 0, 0);
  gemm(0, t1, p1_w2, s1, p1_b2, NTOT, K1, HDIM, 1, 0, 0, 0, 0, 0);
  k_softmax<<<NTOT, 256, 0, stream>>>(s1, K1);
  gemm(0, adj, s1, T1, nullptr, NPG, K1, NPG, NB,
       (long long)NPG * NPG, (long long)NPG * K1, (long long)NPG * K1, 0, 0);
  gemm(1, s1, T1, oa1, nullptr, K1, K1, NPG, NB,
       (long long)NPG * K1, (long long)NPG * K1, (long long)K1 * K1, 0, 0);
  gemm(1, s1, xd, x1, nullptr, K1, HDIM, NPG, NB,
       (long long)NPG * K1, (long long)NPG * HDIM, (long long)K1 * HDIM, 2, 0);   // selu
  gemm(1, s1, s1, ss1, nullptr, K1, K1, NPG, NB,
       (long long)NPG * K1, (long long)NPG * K1, (long long)K1 * K1, 0, 0);
  k_rowsum<<<(int)(((long long)NB * NPG + 3) / 4), 128, 0, stream>>>(adj, degd1, (long long)NB * NPG, NPG);
  k_rowsum<<<(NB + 3) / 4, 128, 0, stream>>>(degd1, twom1, NB, NPG);
  k_ca_colsum<<<(NB * K1 + 255) / 256, 256, 0, stream>>>(s1, degd1, ca1, cs1, NPG, K1);
  k_mat_stats<<<NB, 256, 0, stream>>>(oa1, K1, trA1, trash1);
  k_mat_stats<<<NB, 256, 0, stream>>>(ss1, K1, sstr1, ssq1);
  k_losses<<<(NB + 255) / 256, 256, 0, stream>>>(trA1, ssq1, sstr1, cs1, ca1, twom1,
                                                 K1, (float)NPG, out + NB * NOUT);
  k_finalize_a<<<NB, 256, 0, stream>>>(oa1, a1, K1);

  // ---- dense graph conv 2: x2 = relu(a1 @ x1 @ wrel + brel + x1 @ wroot) ----
  gemm(0, a1, x1, u1, nullptr, K1, HDIM, K1, NB,
       (long long)K1 * K1, (long long)K1 * HDIM, (long long)K1 * HDIM, 0, 0);
  gemm(0, u1, c2_wrel, x2, c2_brel, NB * K1, HDIM, HDIM, 1, 0, 0, 0, 0, 0);
  gemm(0, x1, c2_wroot, x2, nullptr, NB * K1, HDIM, HDIM, 1, 0, 0, 0, 1, 1);      // +=, relu

  // ---- DMoN pool 2 (mask all ones, K1 nodes per graph) ----
  gemm(0, x2, p2_w1, t2, p2_b1, NB * K1, HDIM, HDIM, 1, 0, 0, 0, 0, 0);
  gemm(0, t2, p2_w2, s2, p2_b2, NB * K1, K2, HDIM, 1, 0, 0, 0, 0, 0);
  k_softmax<<<NB * K1, 256, 0, stream>>>(s2, K2);
  gemm(0, a1, s2, T2, nullptr, K1, K2, K1, NB,
       (long long)K1 * K1, (long long)K1 * K2, (long long)K1 * K2, 0, 0);
  gemm(1, s2, T2, oa2, nullptr, K2, K2, K1, NB,
       (long long)K1 * K2, (long long)K1 * K2, (long long)K2 * K2, 0, 0);
  gemm(1, s2, x2, x3, nullptr, K2, HDIM, K1, NB,
       (long long)K1 * K2, (long long)K1 * HDIM, (long long)K2 * HDIM, 2, 0);     // selu
  gemm(1, s2, s2, ss2, nullptr, K2, K2, K1, NB,
       (long long)K1 * K2, (long long)K1 * K2, (long long)K2 * K2, 0, 0);
  k_rowsum<<<(int)(((long long)NB * K1 + 3) / 4), 128, 0, stream>>>(a1, degd2, (long long)NB * K1, K1);
  k_rowsum<<<(NB + 3) / 4, 128, 0, stream>>>(degd2, twom2, NB, K1);
  k_ca_colsum<<<(NB * K2 + 255) / 256, 256, 0, stream>>>(s2, degd2, ca2, cs2, K1, K2);
  k_mat_stats<<<NB, 256, 0, stream>>>(oa2, K2, trA2, trash2);
  k_mat_stats<<<NB, 256, 0, stream>>>(ss2, K2, sstr2, ssq2);
  k_losses<<<(NB + 255) / 256, 256, 0, stream>>>(trA2, ssq2, sstr2, cs2, ca2, twom2,
                                                 K2, (float)K1, out + NB * NOUT);
  k_finalize_a<<<NB, 256, 0, stream>>>(oa2, a2, K2);

  // ---- dense graph conv 3 (no relu) ----
  gemm(0, a2, x3, u2, nullptr, K2, HDIM, K2, NB,
       (long long)K2 * K2, (long long)K2 * HDIM, (long long)K2 * HDIM, 0, 0);
  gemm(0, u2, c3_wrel, x4, c3_brel, NB * K2, HDIM, HDIM, 1, 0, 0, 0, 0, 0);
  gemm(0, x3, c3_wroot, x4, nullptr, NB * K2, HDIM, HDIM, 1, 0, 0, 0, 0, 1);      // +=

  // ---- readout head ----
  k_mean<<<(NB * HDIM + 255) / 256, 256, 0, stream>>>(x4, g, K2);
  gemm(0, g, lin1_w, g1, lin1_b, NB, HDIM, HDIM, 1, 0, 0, 0, 1, 0);               // relu
  k_head<<<(NB + 255) / 256, 256, 0, stream>>>(g1, lin2_w, lin2_b, out);
}
